// CausalSE_39857296507474
// MI455X (gfx1250) — matrix operand order repaired and checked by reading, not compiled, not run on hardware
//
#include <hip/hip_runtime.h>
#include <hip/hip_bf16.h>
#include <stdint.h>

// ---------------- problem constants ----------------
#define BB   16
#define DIM  2048
#define HID  256
#define LL   4096
#define TT   256      // LL / CHUNK
#define CHUNK 16
#define QMAX 7.0f
#define NT   16       // TT/16 n-tiles
#define KT1  64       // DIM/32 k-tiles for GEMM1
#define OT1  16       // HID/16 o-tiles for GEMM1
#define KT2  8        // HID/32 k-tiles for GEMM2
#define OT2  128      // DIM/16 o-tiles for GEMM2

typedef __attribute__((ext_vector_type(16))) _Float16 v16h;
typedef __attribute__((ext_vector_type(8)))  _Float16 v8h;
typedef __attribute__((ext_vector_type(8)))  float    v8f;

// ---------------- 1) chunk mean pool, transposed output [B][T][C] ----------------
__global__ __launch_bounds__(256) void pool_kernel(const float* __restrict__ x,
                                                   float* __restrict__ pooledT) {
  __shared__ float sh[16][17];
  int blk = blockIdx.x;                 // b * 128 * 16 blocks
  int b = blk >> 11, rem = blk & 2047;
  int ctile = rem >> 4, ttile = rem & 15;
  int tid = threadIdx.x;
  int c_local = tid >> 4, part = tid & 15;
  int c = ctile * 16 + c_local;
  int chunk = ttile * 16 + part;
  const float4* px = (const float4*)(x + ((size_t)b * DIM + c) * LL + (size_t)chunk * CHUNK);
  float4 s0 = px[0], s1 = px[1], s2 = px[2], s3 = px[3];
  float s = (s0.x + s0.y + s0.z + s0.w) + (s1.x + s1.y + s1.z + s1.w) +
            (s2.x + s2.y + s2.z + s2.w) + (s3.x + s3.y + s3.z + s3.w);
  sh[c_local][part] = s * (1.0f / 16.0f);
  __syncthreads();
  int t_local = tid >> 4, c_part = tid & 15;
  pooledT[((size_t)b * TT + (ttile * 16 + t_local)) * DIM + ctile * 16 + c_part] =
      sh[c_part][t_local];
}

// ---------------- 2) causal EMA scan; write f16 B-fragments for GEMM1 ----------------
// fragB1 layout: [b][ktile(64)][ntile(16)][lane(32)][e(16)] halfs.
// B-frag mapping (16-bit B 32x16): lane = 16*khalf + n, element e -> k = 16*khalf + e.
__global__ __launch_bounds__(256) void ema_kernel(const float* __restrict__ pooledT,
                                                  const float* __restrict__ ema_r,
                                                  _Float16* __restrict__ fragB1) {
  int tid = blockIdx.x * 256 + threadIdx.x;   // 32768 threads
  int b = tid >> 11, c = tid & (DIM - 1);
  float r = ema_r[c], omr = 1.0f - r;
  int ktile = c >> 5, khalf = (c >> 4) & 1, eb = c & 15;
  size_t base = ((size_t)(b * KT1 + ktile) * NT) * 512 + (size_t)khalf * 256 + eb;
  const float* pin = pooledT + (size_t)b * TT * DIM + c;
  float carry = 0.0f;
  for (int t = 0; t < TT; ++t) {
    carry = r * carry + omr * pin[(size_t)t * DIM];
    fragB1[base + (size_t)(t >> 4) * 512 + (size_t)(t & 15) * 16] = (_Float16)carry;
  }
}

// ---------------- 3) int4 symmetric fake-quant per output row ----------------
__global__ __launch_bounds__(256) void quant_kernel(const float* __restrict__ w,
                                                    _Float16* __restrict__ wq, int C) {
  __shared__ float sm[256];
  int row = blockIdx.x, tid = threadIdx.x;
  const float* wr = w + (size_t)row * C;
  int nper = C >> 8;
  float m = 0.0f;
  for (int j = 0; j < nper; ++j) m = fmaxf(m, fabsf(wr[tid + j * 256]));
  sm[tid] = m;
  __syncthreads();
  for (int s = 128; s > 0; s >>= 1) {
    if (tid < s) sm[tid] = fmaxf(sm[tid], sm[tid + s]);
    __syncthreads();
  }
  float scale = sm[0] * (1.0f / QMAX) + 1e-5f;
  float inv = 1.0f / scale;
  for (int j = 0; j < nper; ++j) {
    float q = rintf(wr[tid + j * 256] * inv);
    q = fminf(QMAX, fmaxf(-QMAX, q));
    wq[(size_t)row * C + tid + j * 256] = (_Float16)(q * scale);
  }
}

// ---------------- 4) pack weights into A-fragment layout ----------------
// A-frag (16-bit A 16x32): lane m = lane&15, half = lane>>4,
// element e -> k = 16*half + e. Pairs (2j,2j+1) are k-consecutive.
__global__ __launch_bounds__(256) void pack_a_kernel(const _Float16* __restrict__ wq,
                                                     _Float16* __restrict__ fragA,
                                                     int C, int KT) {
  int fragIdx = blockIdx.x * 8 + (threadIdx.x >> 5);
  int lane = threadIdx.x & 31;
  int otile = fragIdx / KT, kt = fragIdx - otile * KT;
  int m = lane & 15, half = lane >> 4;
  const _Float16* src = wq + (size_t)(otile * 16 + m) * C + kt * 32;
  uint32_t* dst = (uint32_t*)fragA + ((size_t)fragIdx * 32 + lane) * 8;
#pragma unroll
  for (int j = 0; j < 8; ++j) {
    int e = 2 * j;
    int k = 16 * half + e;
    dst[j] = *(const uint32_t*)(src + k);
  }
}

// ---------------- 5) GEMM1: h1 = relu(W1q @ h + b1), write fragB2 for GEMM2 -------
__global__ __launch_bounds__(256) void gemm1_kernel(const _Float16* __restrict__ fragA,
                                                    const _Float16* __restrict__ fragB,
                                                    const float* __restrict__ b1,
                                                    _Float16* __restrict__ fragB2) {
  int wave = blockIdx.x * 8 + (threadIdx.x >> 5);
  int lane = threadIdx.x & 31;
  int b = wave >> 8, rem = wave & 255;
  int otile = rem >> 4, ntile = rem & 15;
  const v16h* pa = (const v16h*)fragA + (size_t)otile * KT1 * 32 + lane;
  const v16h* pb = (const v16h*)fragB + ((size_t)b * KT1 * NT + ntile) * 32 + lane;
  v8f acc = {};
#pragma unroll 4
  for (int kt = 0; kt < KT1; ++kt) {
    v16h a  = pa[kt * 32];
    v16h bb = pb[kt * (NT * 32)];
    acc = __builtin_amdgcn_wmma_f32_16x16x32_f16(false, a, false, bb, (short)0, acc,
                                                 false, false);
  }
  int half = lane >> 4, n = lane & 15;
  v8h hv;
#pragma unroll
  for (int r2 = 0; r2 < 8; ++r2) {
    int o = otile * 16 + r2 + 8 * half;       // C/D layout: M = r + 8*(lane>>4)
    float v = acc[r2] + b1[o];
    v = v > 0.0f ? v : 0.0f;
    hv[r2] = (_Float16)v;
  }
  // Write directly in GEMM2 B-fragment layout: o -> (ktile2=otile>>1, khalf2=otile&1, e2=m)
  size_t base2 = (((size_t)b * KT2 + (otile >> 1)) * NT + ntile) * 512;
  size_t off = (size_t)((otile & 1) * 16 + n) * 16 + 8 * half;   // 16B aligned
  *(v8h*)(fragB2 + base2 + off) = hv;
}

// ---------------- 6) GEMM2: gate = sigmoid(W2q @ h1 + b2) ----------------
__global__ __launch_bounds__(256) void gemm2_kernel(const _Float16* __restrict__ fragA,
                                                    const _Float16* __restrict__ fragB,
                                                    const float* __restrict__ b2,
                                                    float* __restrict__ gate) {
  int wave = blockIdx.x * 8 + (threadIdx.x >> 5);
  int lane = threadIdx.x & 31;
  int b = wave >> 11, rem = wave & 2047;
  int otile = rem >> 4, ntile = rem & 15;
  const v16h* pa = (const v16h*)fragA + (size_t)otile * KT2 * 32 + lane;
  const v16h* pb = (const v16h*)fragB + ((size_t)b * KT2 * NT + ntile) * 32 + lane;
  v8f acc = {};
#pragma unroll
  for (int kt = 0; kt < KT2; ++kt) {
    v16h a  = pa[kt * 32];
    v16h bb = pb[kt * (NT * 32)];
    acc = __builtin_amdgcn_wmma_f32_16x16x32_f16(false, a, false, bb, (short)0, acc,
                                                 false, false);
  }
  int half = lane >> 4, n = lane & 15;
  int t = ntile * 16 + n;
#pragma unroll
  for (int r2 = 0; r2 < 8; ++r2) {
    int c = otile * 16 + r2 + 8 * half;
    float v = acc[r2] + b2[c];
    gate[((size_t)b * DIM + c) * TT + t] = 1.0f / (1.0f + __expf(-v));
  }
}

// ---------------- 7) upsample gate & multiply residual ----------------
__global__ __launch_bounds__(256) void gatemul_kernel(const float* __restrict__ x,
                                                      const float* __restrict__ gate,
                                                      float* __restrict__ out) {
  size_t i4 = (size_t)blockIdx.x * 256 + threadIdx.x;
  size_t e0 = i4 * 4;
  size_t row = e0 >> 12;          // b*DIM + c   (L = 4096)
  int l = (int)(e0 & (LL - 1));   // 4 elems never cross a 16-chunk boundary
  float g = gate[row * TT + (l >> 4)];
  float4 v = ((const float4*)x)[i4];
  float4 o;
  o.x = v.x * g; o.y = v.y * g; o.z = v.z * g; o.w = v.w * g;
  ((float4*)out)[i4] = o;
}

// ---------------- launcher ----------------
extern "C" void kernel_launch(void* const* d_in, const int* in_sizes, int n_in,
                              void* d_out, int out_size, void* d_ws, size_t ws_size,
                              hipStream_t stream) {
  const float* x     = (const float*)d_in[0];
  const float* w1    = (const float*)d_in[1];
  const float* b1    = (const float*)d_in[2];
  const float* w2    = (const float*)d_in[3];
  const float* b2    = (const float*)d_in[4];
  const float* ema_r = (const float*)d_in[5];
  float* out = (float*)d_out;

  char* ws = (char*)d_ws;
  // pooledT is fully consumed by ema_kernel before gemm2 writes gate -> safe alias.
  float*    pooledT = (float*)ws;                          // 32 MB [B][T][DIM]
  float*    gate    = (float*)ws;                          // 32 MB [B][DIM][T] (alias)
  _Float16* fragB1  = (_Float16*)(ws + (33554432));        // 16 MB
  _Float16* w1q     = (_Float16*)(ws + (33554432 + 16777216));            // 1 MB
  _Float16* w2q     = (_Float16*)(ws + (33554432 + 16777216 + 1048576));  // 1 MB
  _Float16* fragA1  = (_Float16*)(ws + (33554432 + 16777216 + 2 * 1048576)); // 1 MB
  _Float16* fragA2  = (_Float16*)(ws + (33554432 + 16777216 + 3 * 1048576)); // 1 MB
  _Float16* fragB2  = (_Float16*)(ws + (33554432 + 16777216 + 4 * 1048576)); // 2 MB

  // weights: fake-quant then pack to WMMA A-fragments
  quant_kernel<<<HID, 256, 0, stream>>>(w1, w1q, DIM);
  quant_kernel<<<DIM, 256, 0, stream>>>(w2, w2q, HID);
  pack_a_kernel<<<(OT1 * KT1) / 8, 256, 0, stream>>>(w1q, fragA1, DIM, KT1);
  pack_a_kernel<<<(OT2 * KT2) / 8, 256, 0, stream>>>(w2q, fragA2, HID, KT2);

  // pool -> transposed, then causal EMA straight into B-fragments
  pool_kernel<<<BB * (DIM / 16) * (TT / 16), 256, 0, stream>>>(x, pooledT);
  ema_kernel<<<(BB * DIM) / 256, 256, 0, stream>>>(pooledT, ema_r, fragB1);

  // SE bottleneck on WMMA
  gemm1_kernel<<<(BB * OT1 * NT) / 8, 256, 0, stream>>>(fragA1, fragB1, b1, fragB2);
  gemm2_kernel<<<(BB * OT2 * NT) / 8, 256, 0, stream>>>(fragA2, fragB2, b2, gate);

  // gate the residual (bandwidth-bound streaming pass)
  gatemul_kernel<<<(size_t)(BB * DIM * LL / 4) / 256, 256, 0, stream>>>(x, gate, out);
}
